// TorchCausalAttention_34170759807481
// MI455X (gfx1250) — matrix operand order repaired and checked by reading, not compiled, not run on hardware
//
#include <hip/hip_runtime.h>

// ---------------------------------------------------------------------------
// Causal MHA forward for MI455X (gfx1250, wave32, WMMA).
// f16 data / f32 accumulation via v_wmma_f32_16x16x32_f16; softmax in fp32.
// GEMMs use double-buffered GLOBAL_LOAD_ASYNC_TO_LDS_B128 (ASYNCcnt) staging.
// Outputs: out [B,S,D] f32, then attn_weights (head-mean probs) [B,S,S] f32.
// ---------------------------------------------------------------------------

#define B_    2
#define S_    2048
#define D_    1024
#define H_    16
#define HD_   64
#define M_    (B_ * S_)      // 4096 rows of x
#define NQKV_ (3 * D_)       // 3072
#define NEG_INF_ (-1e9f)

typedef __attribute__((ext_vector_type(16))) _Float16 v16h;
typedef __attribute__((ext_vector_type(8)))  _Float16 v8h;
typedef __attribute__((ext_vector_type(8)))  float    v8f;

static __device__ __forceinline__ v16h pack16(v8h lo, v8h hi) {
  v16h r;
#pragma unroll
  for (int i = 0; i < 8; ++i) { r[i] = lo[i]; r[i + 8] = hi[i]; }
  return r;
}

static __device__ __forceinline__ v8f wmma_f16(v16h a, v16h b, v8f c) {
  // (neg_a, A, neg_b, B, c_mod, C, reuse_a, reuse_b)
  return __builtin_amdgcn_wmma_f32_16x16x32_f16(false, a, false, b, (short)0, c,
                                                false, false);
}

static __device__ __forceinline__ v8f zero8f() {
  v8f z = {0.f, 0.f, 0.f, 0.f, 0.f, 0.f, 0.f, 0.f};
  return z;
}

// Async 16B copy global -> LDS, tracked by ASYNCcnt (no VGPR data round-trip).
// lds = byte offset in LDS (low 32 bits of flat shared address).
static __device__ __forceinline__ void g2l_async_b128(unsigned lds,
                                                      const _Float16* g) {
  asm volatile("global_load_async_to_lds_b128 %0, %1, off"
               :
               : "v"(lds), "v"((unsigned long long)(uintptr_t)g)
               : "memory");
}
#define ASYNC_WAIT_3() asm volatile("s_wait_asynccnt 0x3" ::: "memory")
#define ASYNC_WAIT_0() asm volatile("s_wait_asynccnt 0x0" ::: "memory")

// ---------------------------------------------------------------------------
// Elementwise helpers
// ---------------------------------------------------------------------------
__global__ void cast_f32_to_f16(const float* __restrict__ src,
                                _Float16* __restrict__ dst, int n) {
  int i = blockIdx.x * blockDim.x + threadIdx.x;
  if (i < n) dst[i] = (_Float16)src[i];
}

__global__ void zero_f32(float* __restrict__ p, int n) {
  int i = blockIdx.x * blockDim.x + threadIdx.x;
  if (i < n) p[i] = 0.f;
}

// ---------------------------------------------------------------------------
// QKV projection: qkv = x @ Wqkv^T + bqkv, scattered into per-head buffers.
//   qh, kh : [B, H, S, HD]  (f16)
//   vt     : [B, H, HD, S]  (f16, transposed so PV B-fragments are contiguous)
// Block tile 64(M) x 128(N) x 32(K); 256 threads = 8 waves, wave -> 16x64.
// Double-buffered async global->LDS staging.
// ---------------------------------------------------------------------------
__global__ __launch_bounds__(256) void qkv_gemm(
    const _Float16* __restrict__ xh, const _Float16* __restrict__ wh,
    const float* __restrict__ bias, _Float16* __restrict__ qh,
    _Float16* __restrict__ kh, _Float16* __restrict__ vt) {
  __shared__ __align__(16) _Float16 As[2][64][32];
  __shared__ __align__(16) _Float16 Bs[2][128][32];

  const int tid = threadIdx.x;
  const int lane = tid & 31, wave = tid >> 5;
  const int l16 = lane & 15, half16 = lane >> 4;
  const int m_blk = blockIdx.x * 64;
  const int n_blk = blockIdx.y * 128;
  const int m_off = (wave & 3) * 16;
  const int n_off = (wave >> 2) * 64;

  v8f acc[4];
#pragma unroll
  for (int t = 0; t < 4; ++t) acc[t] = zero8f();

  const int a_row = tid >> 2, a_col = (tid & 3) * 8;   // 64 rows x 32 halves
  const int b_row = tid >> 1, b_col = (tid & 1) * 16;  // 128 rows x 32 halves

  const _Float16* gA = xh + (size_t)(m_blk + a_row) * D_ + a_col;
  const _Float16* gB = wh + (size_t)(n_blk + b_row) * D_ + b_col;
  const unsigned ldsA[2] = {(unsigned)(uintptr_t)&As[0][a_row][a_col],
                            (unsigned)(uintptr_t)&As[1][a_row][a_col]};
  const unsigned ldsB[2] = {(unsigned)(uintptr_t)&Bs[0][b_row][b_col],
                            (unsigned)(uintptr_t)&Bs[1][b_row][b_col]};

  // prologue: tile 0 -> buffer 0
  g2l_async_b128(ldsA[0], gA);
  g2l_async_b128(ldsB[0], gB);
  g2l_async_b128(ldsB[0] + 16, gB + 8);

  for (int kt = 0; kt < D_; kt += 32) {
    const int buf = (kt >> 5) & 1;
    if (kt + 32 < D_) {
      g2l_async_b128(ldsA[buf ^ 1], gA + kt + 32);
      g2l_async_b128(ldsB[buf ^ 1], gB + kt + 32);
      g2l_async_b128(ldsB[buf ^ 1] + 16, gB + kt + 40);
      ASYNC_WAIT_3();   // current tile's 3 copies done; next 3 in flight
    } else {
      ASYNC_WAIT_0();
    }
    __syncthreads();

    const int am = m_off + l16, ak = half16 * 16;
    v16h af = pack16(*(const v8h*)&As[buf][am][ak],
                     *(const v8h*)&As[buf][am][ak + 8]);
#pragma unroll
    for (int nt = 0; nt < 4; ++nt) {
      const int bn = n_off + nt * 16 + l16, bk = half16 * 16;
      v16h bf = pack16(*(const v8h*)&Bs[buf][bn][bk],
                       *(const v8h*)&Bs[buf][bn][bk + 8]);
      acc[nt] = wmma_f16(af, bf, acc[nt]);
    }
    __syncthreads();
  }

#pragma unroll
  for (int nt = 0; nt < 4; ++nt) {
    const int gn = n_blk + n_off + nt * 16 + l16;  // [0, 3072)
    const float bv = bias[gn];
    const int sec = gn >> 10;        // 0:q 1:k 2:v
    const int col = gn & 1023;
    const int hh = col >> 6, hd = col & 63;
#pragma unroll
    for (int j = 0; j < 8; ++j) {
      const int gm = m_blk + m_off + j + 8 * half16;  // [0, 4096)
      const int bb = gm >> 11, ss = gm & (S_ - 1);
      const _Float16 hv = (_Float16)(acc[nt][j] + bv);
      const size_t qki = ((size_t)(bb * H_ + hh) * S_ + ss) * HD_ + hd;
      if (sec == 0)
        qh[qki] = hv;
      else if (sec == 1)
        kh[qki] = hv;
      else
        vt[((size_t)(bb * H_ + hh) * HD_ + hd) * S_ + ss] = hv;
    }
  }
}

// ---------------------------------------------------------------------------
// Attention: grid (B*H, S/64), 128 threads = 4 independent waves; each wave
// owns 16 query rows. Two-pass online softmax (pass1 max+sum, pass2 probs,
// atomic head-mean accumulation, and P@V), causal + key-padding mask using
// the reference's -1e9 fill semantics.
// ---------------------------------------------------------------------------
__global__ __launch_bounds__(128) void attn_kernel(
    const _Float16* __restrict__ qh, const _Float16* __restrict__ kh,
    const _Float16* __restrict__ vt, const unsigned char* __restrict__ kpm,
    float* __restrict__ attn, _Float16* __restrict__ ctxh) {
  __shared__ __align__(16) _Float16 pbuf[4][16][32];  // per-wave prob tile

  const int lane = threadIdx.x & 31;
  const int wave = threadIdx.x >> 5;
  const int l16 = lane & 15, half16 = lane >> 4;
  const int bh = blockIdx.x;        // [0, 32)
  const int b = bh >> 4, h = bh & 15;
  const int q0 = blockIdx.y * 64 + wave * 16;

  const _Float16* qbase = qh + (size_t)bh * S_ * HD_;
  const _Float16* kbase = kh + (size_t)bh * S_ * HD_;
  const _Float16* vbase = vt + (size_t)bh * HD_ * S_;
  const unsigned char* pad = kpm + (size_t)b * S_;

  // Q A-fragments (hd split into two K=32 chunks)
  v16h aq0, aq1;
  {
    const _Float16* qr = qbase + (size_t)(q0 + l16) * HD_;
    const int ak = half16 * 16;
    aq0 = pack16(*(const v8h*)(qr + ak), *(const v8h*)(qr + ak + 8));
    aq1 = pack16(*(const v8h*)(qr + 32 + ak), *(const v8h*)(qr + 40 + ak));
  }

  const float scale = 0.125f;  // 1/sqrt(64)
  float mrun[8], srun[8];
#pragma unroll
  for (int j = 0; j < 8; ++j) { mrun[j] = -3.0e38f; srun[j] = 0.f; }

  const int kk = half16 * 16;

  // ---- pass 1: row max + normalizer over the causal key range ----
  const int nkt = q0 / 16 + 1;
  for (int kt = 0; kt < nkt; ++kt) {
    const int kb = kt * 16;
    const _Float16* kr = kbase + (size_t)(kb + l16) * HD_;
    v16h bk0 = pack16(*(const v8h*)(kr + kk), *(const v8h*)(kr + kk + 8));
    v16h bk1 = pack16(*(const v8h*)(kr + 32 + kk), *(const v8h*)(kr + 32 + kk + 8));
    v8f c = zero8f();
    c = wmma_f16(aq0, bk0, c);
    c = wmma_f16(aq1, bk1, c);
#pragma unroll
    for (int j = 0; j < 8; ++j) {
      const int qrow = q0 + j + 8 * half16;
      const int key = kb + l16;
      float sc = c[j] * scale;
      if (key > qrow || !pad[key]) sc = NEG_INF_;
      float tm = sc;
#pragma unroll
      for (int off = 1; off < 16; off <<= 1) tm = fmaxf(tm, __shfl_xor(tm, off, 16));
      const float nm = fmaxf(mrun[j], tm);
      float e = __expf(sc - nm);
#pragma unroll
      for (int off = 1; off < 16; off <<= 1) e += __shfl_xor(e, off, 16);
      srun[j] = srun[j] * __expf(mrun[j] - nm) + e;
      mrun[j] = nm;
    }
  }

  float inv[8];
#pragma unroll
  for (int j = 0; j < 8; ++j) inv[j] = 1.0f / srun[j];

  // ---- pass 2: probs (atomic head-mean) + P@V ----
  v8f cctx[4];
#pragma unroll
  for (int t = 0; t < 4; ++t) cctx[t] = zero8f();

  const int nk2 = q0 / 32 + 1;
  for (int k2 = 0; k2 < nk2; ++k2) {
    const int kb = k2 * 32;
    v16h bk0a, bk0b, bk1a, bk1b;
    {
      const _Float16* kr0 = kbase + (size_t)(kb + l16) * HD_;
      bk0a = pack16(*(const v8h*)(kr0 + kk), *(const v8h*)(kr0 + kk + 8));
      bk0b = pack16(*(const v8h*)(kr0 + 32 + kk), *(const v8h*)(kr0 + 32 + kk + 8));
      const _Float16* kr1 = kbase + (size_t)(kb + 16 + l16) * HD_;
      bk1a = pack16(*(const v8h*)(kr1 + kk), *(const v8h*)(kr1 + kk + 8));
      bk1b = pack16(*(const v8h*)(kr1 + 32 + kk), *(const v8h*)(kr1 + 32 + kk + 8));
    }
    v8f c0 = zero8f(), c1 = zero8f();
    c0 = wmma_f16(aq0, bk0a, c0);
    c0 = wmma_f16(aq1, bk0b, c0);
    c1 = wmma_f16(aq0, bk1a, c1);
    c1 = wmma_f16(aq1, bk1b, c1);

#pragma unroll
    for (int j = 0; j < 8; ++j) {
      const int qrow = q0 + j + 8 * half16;
      const int key0 = kb + l16, key1 = kb + 16 + l16;
      float s0 = c0[j] * scale, s1 = c1[j] * scale;
      if (key0 > qrow || !pad[key0]) s0 = NEG_INF_;
      if (key1 > qrow || !pad[key1]) s1 = NEG_INF_;
      const float p0 = __expf(s0 - mrun[j]) * inv[j];
      const float p1 = __expf(s1 - mrun[j]) * inv[j];
      float* arow = attn + ((size_t)(b * S_ + qrow)) * S_;
      if (p0 != 0.f) atomicAdd(arow + key0, p0 * (1.0f / H_));
      if (p1 != 0.f) atomicAdd(arow + key1, p1 * (1.0f / H_));
      pbuf[wave][j + 8 * half16][l16] = (_Float16)p0;
      pbuf[wave][j + 8 * half16][16 + l16] = (_Float16)p1;
    }

    // C-layout -> A-fragment layout through LDS (same-wave, DS in-order)
    const int ak = half16 * 16;
    v16h ap = pack16(*(const v8h*)&pbuf[wave][l16][ak],
                     *(const v8h*)&pbuf[wave][l16][ak + 8]);
#pragma unroll
    for (int t = 0; t < 4; ++t) {
      const _Float16* vr = vbase + (size_t)(t * 16 + l16) * S_ + kb + half16 * 16;
      v16h bv = pack16(*(const v8h*)vr, *(const v8h*)(vr + 8));
      cctx[t] = wmma_f16(ap, bv, cctx[t]);
    }
  }

  // write ctx [B, S, D] f16 (heads re-interleaved)
#pragma unroll
  for (int t = 0; t < 4; ++t)
#pragma unroll
    for (int j = 0; j < 8; ++j) {
      const int srow = q0 + j + 8 * half16;
      const int col = h * HD_ + t * 16 + l16;
      ctxh[((size_t)b * S_ + srow) * D_ + col] = (_Float16)cctx[t][j];
    }
}

// ---------------------------------------------------------------------------
// Output projection: out = ctx @ Wout^T + bout  (fp32 result)
// ---------------------------------------------------------------------------
__global__ __launch_bounds__(256) void out_gemm(
    const _Float16* __restrict__ ctxh, const _Float16* __restrict__ wh,
    const float* __restrict__ bias, float* __restrict__ out) {
  __shared__ __align__(16) _Float16 As[2][64][32];
  __shared__ __align__(16) _Float16 Bs[2][128][32];

  const int tid = threadIdx.x;
  const int lane = tid & 31, wave = tid >> 5;
  const int l16 = lane & 15, half16 = lane >> 4;
  const int m_blk = blockIdx.x * 64;
  const int n_blk = blockIdx.y * 128;
  const int m_off = (wave & 3) * 16;
  const int n_off = (wave >> 2) * 64;

  v8f acc[4];
#pragma unroll
  for (int t = 0; t < 4; ++t) acc[t] = zero8f();

  const int a_row = tid >> 2, a_col = (tid & 3) * 8;
  const int b_row = tid >> 1, b_col = (tid & 1) * 16;

  const _Float16* gA = ctxh + (size_t)(m_blk + a_row) * D_ + a_col;
  const _Float16* gB = wh + (size_t)(n_blk + b_row) * D_ + b_col;
  const unsigned ldsA[2] = {(unsigned)(uintptr_t)&As[0][a_row][a_col],
                            (unsigned)(uintptr_t)&As[1][a_row][a_col]};
  const unsigned ldsB[2] = {(unsigned)(uintptr_t)&Bs[0][b_row][b_col],
                            (unsigned)(uintptr_t)&Bs[1][b_row][b_col]};

  g2l_async_b128(ldsA[0], gA);
  g2l_async_b128(ldsB[0], gB);
  g2l_async_b128(ldsB[0] + 16, gB + 8);

  for (int kt = 0; kt < D_; kt += 32) {
    const int buf = (kt >> 5) & 1;
    if (kt + 32 < D_) {
      g2l_async_b128(ldsA[buf ^ 1], gA + kt + 32);
      g2l_async_b128(ldsB[buf ^ 1], gB + kt + 32);
      g2l_async_b128(ldsB[buf ^ 1] + 16, gB + kt + 40);
      ASYNC_WAIT_3();
    } else {
      ASYNC_WAIT_0();
    }
    __syncthreads();

    const int am = m_off + l16, ak = half16 * 16;
    v16h af = pack16(*(const v8h*)&As[buf][am][ak],
                     *(const v8h*)&As[buf][am][ak + 8]);
#pragma unroll
    for (int nt = 0; nt < 4; ++nt) {
      const int bn = n_off + nt * 16 + l16, bk = half16 * 16;
      v16h bf = pack16(*(const v8h*)&Bs[buf][bn][bk],
                       *(const v8h*)&Bs[buf][bn][bk + 8]);
      acc[nt] = wmma_f16(af, bf, acc[nt]);
    }
    __syncthreads();
  }

#pragma unroll
  for (int nt = 0; nt < 4; ++nt) {
    const int gn = n_blk + n_off + nt * 16 + l16;
    const float bv = bias[gn];
#pragma unroll
    for (int j = 0; j < 8; ++j) {
      const int gm = m_blk + m_off + j + 8 * half16;
      out[(size_t)gm * D_ + gn] = acc[nt][j] + bv;
    }
  }
}

// ---------------------------------------------------------------------------
// Launch
// ---------------------------------------------------------------------------
extern "C" void kernel_launch(void* const* d_in, const int* in_sizes, int n_in,
                              void* d_out, int out_size, void* d_ws,
                              size_t ws_size, hipStream_t stream) {
  (void)in_sizes; (void)n_in; (void)out_size; (void)ws_size;

  const float* x = (const float*)d_in[0];
  const float* Wqkv = (const float*)d_in[1];
  const float* bqkv = (const float*)d_in[2];
  const float* Wout = (const float*)d_in[3];
  const float* bout = (const float*)d_in[4];
  const unsigned char* kpm = (const unsigned char*)d_in[5];  // bool [B,S]

  float* out = (float*)d_out;                    // [B,S,D]  = 4,194,304 f32
  float* attn = out + (size_t)M_ * D_;           // [B,S,S]  = 8,388,608 f32

  // workspace layout (f16 buffers), total 48 MB
  char* ws = (char*)d_ws;
  _Float16* xh    = (_Float16*)(ws);                          // 8 MB
  _Float16* wqkvh = (_Float16*)(ws + (8u << 20));             // 6 MB
  _Float16* wouth = (_Float16*)(ws + (14u << 20));            // 2 MB
  _Float16* qh    = (_Float16*)(ws + (16u << 20));            // 8 MB
  _Float16* kh    = (_Float16*)(ws + (24u << 20));            // 8 MB
  _Float16* vt    = (_Float16*)(ws + (32u << 20));            // 8 MB
  _Float16* ctxh  = (_Float16*)(ws + (40u << 20));            // 8 MB

  const int TPB = 256;
  cast_f32_to_f16<<<(M_ * D_) / TPB, TPB, 0, stream>>>(x, xh, M_ * D_);
  cast_f32_to_f16<<<(NQKV_ * D_) / TPB, TPB, 0, stream>>>(Wqkv, wqkvh, NQKV_ * D_);
  cast_f32_to_f16<<<(D_ * D_) / TPB, TPB, 0, stream>>>(Wout, wouth, D_ * D_);
  zero_f32<<<(B_ * S_ * S_) / TPB, TPB, 0, stream>>>(attn, B_ * S_ * S_);

  dim3 gq(M_ / 64, NQKV_ / 128);
  qkv_gemm<<<gq, 256, 0, stream>>>(xh, wqkvh, bqkv, qh, kh, vt);

  dim3 ga(B_ * H_, S_ / 64);
  attn_kernel<<<ga, 128, 0, stream>>>(qh, kh, vt, kpm, attn, ctxh);

  dim3 go(M_ / 64, D_ / 128);
  out_gemm<<<go, 256, 0, stream>>>(ctxh, wouth, bout, out);
}
